// S3Mamba_Pan_38465727103447
// MI455X (gfx1250) — compile-verified, matrix operand order checked by reading, not run on hardware
//
#include <hip/hip_runtime.h>
#include <hip/hip_bf16.h>

// ---------------------------------------------------------------------------
// CDNA5 / gfx1250 implementation of the S3-Mamba pansharpening forward pass.
// All tensors between DWT and IDWT are token-major (L x C), so every GEMM
// operand is contiguous row-major. The WMMA GEMM is software-pipelined:
// the next K-block's raw float4s are loaded while the current block is
// converted to bf16 and fed to 4 back-to-back v_wmma_f32_16x16x32_bf16.
// __launch_bounds__(32,1) lifts the VGPR budget so both buffer sets stay
// in registers.
// ---------------------------------------------------------------------------

typedef __attribute__((ext_vector_type(16))) __bf16 v16bf;
typedef __attribute__((ext_vector_type(8)))  float  v8f;

__device__ __forceinline__ float silu_f(float x)     { return x / (1.f + __expf(-x)); }
__device__ __forceinline__ float sigmoid_f(float x)  { return 1.f / (1.f + __expf(-x)); }
__device__ __forceinline__ float softplus_f(float x) { return (x > 20.f) ? x : log1pf(__expf(x)); }
__device__ __forceinline__ float leaky_f(float x)    { return x > 0.f ? x : 0.1f * x; }

__device__ __forceinline__ v16bf cvt_frag(float4 a, float4 b, float4 c, float4 d)
{
    v16bf f;
    f[0]  = (__bf16)a.x; f[1]  = (__bf16)a.y; f[2]  = (__bf16)a.z; f[3]  = (__bf16)a.w;
    f[4]  = (__bf16)b.x; f[5]  = (__bf16)b.y; f[6]  = (__bf16)b.z; f[7]  = (__bf16)b.w;
    f[8]  = (__bf16)c.x; f[9]  = (__bf16)c.y; f[10] = (__bf16)c.z; f[11] = (__bf16)c.w;
    f[12] = (__bf16)d.x; f[13] = (__bf16)d.y; f[14] = (__bf16)d.z; f[15] = (__bf16)d.w;
    return f;
}

// Raw data of one 16(M)x64(N)x32(K) block: A fragment + 4 B fragments.
// The fragment's K pattern is two contiguous 8-float runs at kb and kb+16,
// all 16B-aligned (leading dims are multiples of 4, kb a multiple of 8).
struct RawBlk {
    float4 a[4];
    float4 b[4][4];
};

template <bool HAS_AP>
__device__ __forceinline__ void load_blk(const float* __restrict__ Arow,
                                         const float* const Wrow[4],
                                         const float* __restrict__ a_plus,
                                         int kb, RawBlk& r)
{
    r.a[0] = *(const float4*)(Arow + kb);
    r.a[1] = *(const float4*)(Arow + kb + 4);
    r.a[2] = *(const float4*)(Arow + kb + 16);
    r.a[3] = *(const float4*)(Arow + kb + 20);
#pragma unroll
    for (int t = 0; t < 4; ++t) {
        r.b[t][0] = *(const float4*)(Wrow[t] + kb);
        r.b[t][1] = *(const float4*)(Wrow[t] + kb + 4);
        r.b[t][2] = *(const float4*)(Wrow[t] + kb + 16);
        r.b[t][3] = *(const float4*)(Wrow[t] + kb + 20);
    }
    if (HAS_AP) {
        float4 p0 = *(const float4*)(a_plus + kb);
        float4 p1 = *(const float4*)(a_plus + kb + 4);
        float4 p2 = *(const float4*)(a_plus + kb + 16);
        float4 p3 = *(const float4*)(a_plus + kb + 20);
        r.a[0].x += p0.x; r.a[0].y += p0.y; r.a[0].z += p0.z; r.a[0].w += p0.w;
        r.a[1].x += p1.x; r.a[1].y += p1.y; r.a[1].z += p1.z; r.a[1].w += p1.w;
        r.a[2].x += p2.x; r.a[2].y += p2.y; r.a[2].z += p2.z; r.a[2].w += p2.w;
        r.a[3].x += p3.x; r.a[3].y += p3.y; r.a[3].z += p3.z; r.a[3].w += p3.w;
    }
}

__device__ __forceinline__ void compute_blk(const RawBlk& r, v8f acc[4])
{
    const v16bf af = cvt_frag(r.a[0], r.a[1], r.a[2], r.a[3]);
#pragma unroll
    for (int t = 0; t < 4; ++t) {
        const v16bf bf = cvt_frag(r.b[t][0], r.b[t][1], r.b[t][2], r.b[t][3]);
        acc[t] = __builtin_amdgcn_wmma_f32_16x16x32_bf16(
                     false, af, false, bf, (short)0, acc[t], false, false);
    }
}

// ---------------------------------------------------------------------------
// WMMA GEMM, all operands row(token)-major with runtime leading dims:
//   Out[m,n] = act( sum_k (A[m,k]+a_plus[k]) * W[n, w_off+k] + bias[n] + Cin[m,n] )
// Grid: (ceil(N/64), M/16); block: 32 (one wave -> 16x64 tile).
// M % 16 == 0 at every call site (M = 4096), N columns clamp-loaded.
// ---------------------------------------------------------------------------
template <bool HAS_AP>
__global__ void __launch_bounds__(32, 1) wmma_gemm_kernel(
    const float* __restrict__ A, int a_rs, const float* __restrict__ a_plus,
    const float* __restrict__ W, int w_ld, int w_off,
    const float* __restrict__ bias,
    const float* __restrict__ Cin, int c_rs,
    float* __restrict__ Out, int o_rs,
    int N, int K, int act)
{
    const int lane   = threadIdx.x & 31;
    const int tn0    = blockIdx.x * 64;
    const int tm     = blockIdx.y * 16;
    const int col    = lane & 15;       // N index within tile (B/C/D operands)
    const int half   = lane >> 4;       // K-group select for A/B fragments
    const int kb_off = half ? 8 : 0;

    v8f acc[4];
#pragma unroll
    for (int t = 0; t < 4; ++t) {
        const int n   = tn0 + 16 * t + col;
        const int ncl = (n < N) ? n : (N - 1);
        if (Cin) {
#pragma unroll
            for (int i = 0; i < 8; ++i)   // C/D: vgpr i -> M=i (lanes 0-15) / M=i+8
                acc[t][i] = Cin[(long)(tm + i + 8 * half) * c_rs + ncl];
        } else {
#pragma unroll
            for (int i = 0; i < 8; ++i) acc[t][i] = 0.f;
        }
    }

    const float* Arow = A + (long)(tm + col) * a_rs;   // A fragment: lane -> M row
    const float* Wrow[4];
#pragma unroll
    for (int t = 0; t < 4; ++t) {
        const int n   = tn0 + 16 * t + col;
        const int ncl = (n < N) ? n : (N - 1);
        Wrow[t] = W + (long)ncl * w_ld + w_off;
    }

    const int kfull = K & ~31;
    if (kfull > 0) {
        // software pipeline: prefetch block i+1 while computing block i
        RawBlk cur, nxt;
        load_blk<HAS_AP>(Arow, Wrow, a_plus, kb_off, cur);
        for (int k0 = 32; k0 < kfull; k0 += 32) {
            load_blk<HAS_AP>(Arow, Wrow, a_plus, k0 + kb_off, nxt);
            compute_blk(cur, acc);
            cur = nxt;
        }
        compute_blk(cur, acc);
    }

    if (kfull < K) {   // guarded K tail (only K=4 / K=12 call sites reach this)
        const int kb = kfull + kb_off;
        v16bf af;
#pragma unroll
        for (int i = 0; i < 8; ++i) {
            const int kk = kb + ((i < 4) ? 2 * i : 16 + 2 * (i - 4));
            float f0 = 0.f, f1 = 0.f;
            if (kk < K)     { f0 = Arow[kk];     if (HAS_AP) f0 += a_plus[kk]; }
            if (kk + 1 < K) { f1 = Arow[kk + 1]; if (HAS_AP) f1 += a_plus[kk + 1]; }
            af[2 * i]     = (__bf16)f0;
            af[2 * i + 1] = (__bf16)f1;
        }
#pragma unroll
        for (int t = 0; t < 4; ++t) {
            v16bf bf;
#pragma unroll
            for (int i = 0; i < 8; ++i) {
                const int kk = kb + ((i < 4) ? 2 * i : 16 + 2 * (i - 4));
                float g0 = (kk < K)     ? Wrow[t][kk]     : 0.f;
                float g1 = (kk + 1 < K) ? Wrow[t][kk + 1] : 0.f;
                bf[2 * i]     = (__bf16)g0;
                bf[2 * i + 1] = (__bf16)g1;
            }
            acc[t] = __builtin_amdgcn_wmma_f32_16x16x32_bf16(
                         false, af, false, bf, (short)0, acc[t], false, false);
        }
    }

#pragma unroll
    for (int t = 0; t < 4; ++t) {
        const int n = tn0 + 16 * t + col;
        if (n < N) {
            const float bv = bias ? bias[n] : 0.f;
#pragma unroll
            for (int i = 0; i < 8; ++i) {
                float v = acc[t][i] + bv;
                if (act == 1)      v = silu_f(v);
                else if (act == 2) v = leaky_f(v);
                Out[(long)(tm + i + 8 * half) * o_rs + n] = v;
            }
        }
    }
}

// ---------------------------------------------------------------------------
// Direct 3x3 conv (pad=1), optional leaky-relu and residual add. NCHW planes.
// ---------------------------------------------------------------------------
__global__ void conv3x3_kernel(const float* __restrict__ in, const float* __restrict__ w,
                               const float* __restrict__ b, const float* __restrict__ resid,
                               float* __restrict__ out, int Cin, int Cout, int H, int W, int act)
{
    long idx = (long)blockIdx.x * blockDim.x + threadIdx.x;
    long total = (long)Cout * H * W;
    if (idx >= total) return;
    int x = idx % W, y = (idx / W) % H, co = idx / ((long)W * H);
    float acc = b ? b[co] : 0.f;
    for (int ci = 0; ci < Cin; ++ci) {
        const float* ip = in + (long)ci * H * W;
        const float* wp = w + ((long)co * Cin + ci) * 9;
#pragma unroll
        for (int ky = 0; ky < 3; ++ky) {
            int yy = y + ky - 1;
            if (yy < 0 || yy >= H) continue;
#pragma unroll
            for (int kx = 0; kx < 3; ++kx) {
                int xx = x + kx - 1;
                if (xx < 0 || xx >= W) continue;
                acc += wp[ky * 3 + kx] * ip[(long)yy * W + xx];
            }
        }
    }
    if (act == 1) acc = leaky_f(acc);
    if (resid) acc += resid[idx];
    out[idx] = acc;
}

// ---------------------------------------------------------------------------
// Haar DWT: NCHW planes -> token-major (L, 4C) rows [ll | lh | hl | hh].
// ---------------------------------------------------------------------------
__global__ void dwt_kernel(const float* __restrict__ in, float* __restrict__ out,
                           int C, int H, int W)
{
    int H2 = H / 2, W2 = W / 2;
    long idx = (long)blockIdx.x * blockDim.x + threadIdx.x;
    long total = (long)C * H2 * W2;
    if (idx >= total) return;
    int x = idx % W2, y = (idx / W2) % H2, c = idx / ((long)W2 * H2);
    const float* ip = in + (long)c * H * W;
    float a  = ip[(long)(2 * y) * W + 2 * x];
    float bb = ip[(long)(2 * y) * W + 2 * x + 1];
    float cc = ip[(long)(2 * y + 1) * W + 2 * x];
    float dd = ip[(long)(2 * y + 1) * W + 2 * x + 1];
    long row = ((long)y * W2 + x) * (4L * C);
    out[row + c]         = 0.25f * (a + bb + cc + dd);
    out[row + C + c]     = 0.25f * (a + bb - cc - dd);
    out[row + 2 * C + c] = 0.25f * (a - bb + cc - dd);
    out[row + 3 * C + c] = 0.25f * (a - bb - cc + dd);
}

// ---------------------------------------------------------------------------
// Haar IDWT: token-major (L, 4C) -> NCHW planes (2H2 x 2W2).
// ---------------------------------------------------------------------------
__global__ void idwt_kernel(const float* __restrict__ in, float* __restrict__ out,
                            int C, int H2, int W2)
{
    long idx = (long)blockIdx.x * blockDim.x + threadIdx.x;
    long total = (long)C * H2 * W2;
    if (idx >= total) return;
    int x = idx % W2, y = (idx / W2) % H2, c = idx / ((long)W2 * H2);
    long row = ((long)y * W2 + x) * (4L * C);
    float ll = in[row + c];
    float lh = in[row + C + c];
    float hl = in[row + 2 * C + c];
    float hh = in[row + 3 * C + c];
    long plane = (long)H2 * W2;
    float* op = out + (long)c * 4 * plane;
    int W = 2 * W2;
    op[(long)(2 * y) * W + 2 * x]         = ll + lh + hl + hh;
    op[(long)(2 * y) * W + 2 * x + 1]     = ll + lh - hl - hh;
    op[(long)(2 * y + 1) * W + 2 * x]     = ll - lh + hl - hh;
    op[(long)(2 * y + 1) * W + 2 * x + 1] = ll - lh - hl + hh;
}

// ---------------------------------------------------------------------------
// Depthwise causal conv1d (K=4) + SiLU. Token-major in (L x ldx) / out (L x d).
// ---------------------------------------------------------------------------
__global__ void dwconv_silu_kernel(const float* __restrict__ X, int ldx,
                                   const float* __restrict__ w, const float* __restrict__ b,
                                   float* __restrict__ out, int d, long L)
{
    long idx = (long)blockIdx.x * blockDim.x + threadIdx.x;
    if (idx >= (long)d * L) return;
    int c = (int)(idx % d);
    long l = idx / d;
    float acc = b[c];
#pragma unroll
    for (int j = 0; j < 4; ++j) {
        long li = l - 3 + j;
        if (li >= 0) acc += w[c * 4 + j] * X[li * ldx + c];
    }
    out[l * d + c] = silu_f(acc);
}

// ---------------------------------------------------------------------------
// LayerNorm over channels. Token-major in (L x ld_in) -> out (L x C).
// ---------------------------------------------------------------------------
__global__ void ln_kernel(const float* __restrict__ in, int ld_in,
                          const float* __restrict__ g, const float* __restrict__ b,
                          float* __restrict__ out, int C, long L)
{
    long l = (long)blockIdx.x * blockDim.x + threadIdx.x;
    if (l >= L) return;
    const float* row = in + l * ld_in;
    float m = 0.f, q = 0.f;
    for (int c = 0; c < C; ++c) { float v = row[c]; m += v; q += v * v; }
    m /= C;
    float var = q / C - m * m;
    float inv = rsqrtf(var + 1e-5f);
    for (int c = 0; c < C; ++c)
        out[l * C + c] = (row[c] - m) * inv * g[c] + b[c];
}

// ---------------------------------------------------------------------------
// Selective scan: one thread per channel, 16-state recurrence in registers.
// All per-step tensors token-major -> lane-coalesced loads/stores each step.
// ---------------------------------------------------------------------------
__global__ void scan_kernel(const float* __restrict__ u,      // (L, d)
                            const float* __restrict__ delta,  // (L, d)
                            const float* __restrict__ A_log,
                            const float* __restrict__ Bm, int bld,
                            const float* __restrict__ Cm, int cld,
                            const float* __restrict__ Dv,
                            const float* __restrict__ zg, int zld,
                            const float* __restrict__ dtb,
                            float* __restrict__ y,            // (L, d)
                            int d, long L)
{
    int ch = blockIdx.x * blockDim.x + threadIdx.x;
    if (ch >= d) return;
    float A[16], h[16];
#pragma unroll
    for (int n = 0; n < 16; ++n) { A[n] = -__expf(A_log[ch * 16 + n]); h[n] = 0.f; }
    const float db = dtb[ch];
    const float Dc = Dv[ch];
    for (long l = 0; l < L; ++l) {
        float dl = softplus_f(delta[l * d + ch] + db);
        float uv = u[l * d + ch];
        float acc = 0.f;
#pragma unroll
        for (int n = 0; n < 16; ++n) {
            float bn = Bm[l * bld + n];
            float cn = Cm[l * cld + n];
            h[n] = h[n] * __expf(dl * A[n]) + dl * bn * uv;
            acc += h[n] * cn;
        }
        float zv = zg[l * zld + ch];
        y[l * d + ch] = (acc + Dc * uv) * silu_f(zv);
    }
}

// ---------------------------------------------------------------------------
// GSAM helpers.
// ---------------------------------------------------------------------------
__global__ void avgpool_kernel(const float* __restrict__ in, float* __restrict__ out, int HW)
{
    __shared__ float s[256];
    int c = blockIdx.x;
    float acc = 0.f;
    for (int i = threadIdx.x; i < HW; i += 256) acc += in[(long)c * HW + i];
    s[threadIdx.x] = acc;
    __syncthreads();
    for (int st = 128; st > 0; st >>= 1) {
        if ((int)threadIdx.x < st) s[threadIdx.x] += s[threadIdx.x + st];
        __syncthreads();
    }
    if (threadIdx.x == 0) out[c] = s[0] / (float)HW;
}

__global__ void dense_small_kernel(const float* __restrict__ in, const float* __restrict__ w,
                                   const float* __restrict__ b, float* __restrict__ out,
                                   int K, int N, int act)
{
    int o = blockIdx.x * blockDim.x + threadIdx.x;
    if (o >= N) return;
    float acc = b ? b[o] : 0.f;
    for (int k = 0; k < K; ++k) acc += w[(long)o * K + k] * in[k];
    if (act == 1) acc = leaky_f(acc);
    out[o] = acc;
}

// ---------------------------------------------------------------------------
// ADR: channel statistics, 1D conv gate, channel scaling (NCHW planes).
// ---------------------------------------------------------------------------
__global__ void chstats_kernel(const float* __restrict__ in, float* __restrict__ y,
                               int C, int HW)
{
    __shared__ float ss[256], sa[256], sq[256];
    int c = blockIdx.x;
    float s = 0.f, a = 0.f, q = 0.f;
    for (int i = threadIdx.x; i < HW; i += 256) {
        float v = in[(long)c * HW + i];
        s += v; a += fabsf(v); q += v * v;
    }
    ss[threadIdx.x] = s; sa[threadIdx.x] = a; sq[threadIdx.x] = q;
    __syncthreads();
    for (int st = 128; st > 0; st >>= 1) {
        if ((int)threadIdx.x < st) {
            ss[threadIdx.x] += ss[threadIdx.x + st];
            sa[threadIdx.x] += sa[threadIdx.x + st];
            sq[threadIdx.x] += sq[threadIdx.x + st];
        }
        __syncthreads();
    }
    if (threadIdx.x == 0) {
        float mean = ss[0] / HW;
        float var  = (sq[0] - HW * mean * mean) / (HW - 1);   // ddof=1
        y[c]         = sa[0] / HW;
        y[C + c]     = mean;
        y[2 * C + c] = sqrtf(var + 1e-6f);
    }
}

__global__ void adr_gate_kernel(const float* __restrict__ y, const float* __restrict__ w,
                                float* __restrict__ g, int C)
{
    int c = blockIdx.x * blockDim.x + threadIdx.x;
    if (c >= C) return;
    float acc = 0.f;
    for (int ic = 0; ic < 3; ++ic)
        for (int k = 0; k < 3; ++k) {
            int p = c + k - 1;
            if (p >= 0 && p < C) acc += y[ic * C + p] * w[ic * 3 + k];
        }
    g[c] = sigmoid_f(acc);
}

__global__ void scale_kernel(float* __restrict__ x, const float* __restrict__ g,
                             int C, int HW)
{
    long idx = (long)blockIdx.x * blockDim.x + threadIdx.x;
    if (idx >= (long)C * HW) return;
    x[idx] *= g[idx / HW];
}

// ---------------------------------------------------------------------------
// Host side
// ---------------------------------------------------------------------------
static inline void launch_gemm(hipStream_t s,
    const float* A, int ars, const float* aplus,
    const float* W, int wld, int woff, const float* bias,
    const float* Cin, int crs, float* Out, int ors,
    int M, int N, int K, int act)
{
    dim3 g((N + 63) / 64, M / 16), bl(32);
    if (aplus)
        hipLaunchKernelGGL(wmma_gemm_kernel<true>, g, bl, 0, s,
                           A, ars, aplus, W, wld, woff, bias, Cin, crs, Out, ors, N, K, act);
    else
        hipLaunchKernelGGL(wmma_gemm_kernel<false>, g, bl, 0, s,
                           A, ars, (const float*)nullptr, W, wld, woff, bias, Cin, crs, Out, ors, N, K, act);
}

static inline dim3 grid1d(long n, int bs) { return dim3((unsigned)((n + bs - 1) / bs)); }

extern "C" void kernel_launch(void* const* d_in, const int* in_sizes, int n_in,
                              void* d_out, int out_size, void* d_ws, size_t ws_size,
                              hipStream_t stream)
{
    (void)in_sizes; (void)n_in; (void)out_size; (void)ws_size;
    const int H = 128, W = 128, HW = H * W;
    const int H2 = 64, W2 = 64;
    const long L = (long)H2 * W2;           // 4096
    const int DIM = 64, DIM3 = 192;
    const int D1 = 128, R1 = 4;             // SSS inner dim / rank
    const int D2 = 384, R2 = 12;            // STS inner dim / rank

#define IN(i) ((const float*)d_in[(i)])
    const float* ms  = IN(0);
    const float* pan = IN(1);

    // ---- workspace layout (floats); total ~60 MB ----
    float* F = (float*)d_ws;
    long off = 0;
    long o_md   = off; off += 256L * L;     // m_d token-major (L, 256): cols [m_l(0:64) | m_h(64:256)]
    long o_pd   = off; off += 256L * L;     // p_d token-major (L, 256): cols [p_l | p_h]
    long o_kex  = off; off += 192L * L;     // token-major (L, 192)
    long o_z    = off; off += 64;
    long o_zmid = off; off += 256;
    long o_pool = off; off += 64;
    long o_adry = off; off += 256;          // 3 x 64 (padded)
    long o_adrg = off; off += 64;
    long S = off;
    // phase A (pre-block) scratch
    long o_fm = S, o_fp = S + 1048576, o_h1 = S + 2097152, o_h2 = S + 2621440;
    // phase B: SSS scratch (consumed before STS scratch reuses the region)
    long o_in1 = S, o_xc1 = S + 1048576, o_xd1 = S + 1572864,
         o_dl1 = S + 1720320, o_y1 = S + 2244608;
    // phase B: STS scratch
    long o_in2 = S, o_xc2 = S + 1572864, o_lnb = S + 3145728, o_kp = S + 3932160,
         o_qp = S + 5505024, o_xd2 = S + 7077888, o_gt = S + 7274496,
         o_dl2 = S + 8847360, o_y2 = S + 10420224;
    // phase C
    long o_idwt = S;

    float* m_d = F + o_md;  float* m_l = m_d;  float* m_h = m_d + 64;   // column offsets, rs=256
    float* p_d = F + o_pd;  float* p_h = p_d + 64;
    float* kex  = F + o_kex;
    float* zv   = F + o_z;
    float* zmid = F + o_zmid;
    float* pool = F + o_pool;
    float* adry = F + o_adry;
    float* adrg = F + o_adrg;

    // ================= GSAM -> z =================
    hipLaunchKernelGGL(conv3x3_kernel, grid1d(32L * HW, 256), dim3(256), 0, stream,
                       ms, IN(6), IN(7), (const float*)nullptr, F + o_h1, 8, 32, H, W, 1);
    hipLaunchKernelGGL(conv3x3_kernel, grid1d(64L * HW, 256), dim3(256), 0, stream,
                       F + o_h1, IN(8), IN(9), (const float*)nullptr, F + o_h2, 32, 64, H, W, 1);
    hipLaunchKernelGGL(avgpool_kernel, dim3(64), dim3(256), 0, stream, F + o_h2, pool, HW);
    hipLaunchKernelGGL(dense_small_kernel, dim3(1), dim3(256), 0, stream,
                       pool, IN(10), IN(11), zmid, 64, 256, 1);
    hipLaunchKernelGGL(dense_small_kernel, dim3(1), dim3(64), 0, stream,
                       zmid, IN(12), IN(13), zv, 256, 64, 0);

    // ================= stem convs + DWT =================
    hipLaunchKernelGGL(conv3x3_kernel, grid1d(64L * HW, 256), dim3(256), 0, stream,
                       ms, IN(2), IN(3), (const float*)nullptr, F + o_fm, 8, 64, H, W, 0);
    hipLaunchKernelGGL(conv3x3_kernel, grid1d(64L * HW, 256), dim3(256), 0, stream,
                       pan, IN(4), IN(5), (const float*)nullptr, F + o_fp, 1, 64, H, W, 0);
    hipLaunchKernelGGL(dwt_kernel, grid1d(64L * L, 256), dim3(256), 0, stream,
                       F + o_fm, m_d, 64, H, W);
    hipLaunchKernelGGL(dwt_kernel, grid1d(64L * L, 256), dim3(256), 0, stream,
                       F + o_fp, p_d, 64, H, W);

    // k_ex = 1x1 conv (p_l: 64 -> 192): A = p_d rows (cols 0..63), token-major out
    launch_gemm(stream, p_d, 256, nullptr, IN(14), 64, 0, IN(15),
                nullptr, 0, kex, 192, (int)L, 192, 64, 0);

    // ================= 3 layers =================
    for (int blk = 0; blk < 3; ++blk) {
        const int bb = 16 + 28 * blk;   // sss params
        const int sb = bb + 9;          // sts params

        // -------- SSS block (dim=64, d=128, r=4) --------
        launch_gemm(stream, m_l, 256, zv, IN(bb + 0), DIM, 0, nullptr,
                    nullptr, 0, F + o_in1, 2 * D1, (int)L, 2 * D1, DIM, 0);
        hipLaunchKernelGGL(dwconv_silu_kernel, grid1d((long)D1 * L, 256), dim3(256), 0, stream,
                           F + o_in1, 2 * D1, IN(bb + 1), IN(bb + 2), F + o_xc1, D1, L);
        launch_gemm(stream, F + o_xc1, D1, nullptr, IN(bb + 3), D1, 0, nullptr,
                    nullptr, 0, F + o_xd1, R1 + 32, (int)L, R1 + 32, D1, 0);
        launch_gemm(stream, F + o_xd1, R1 + 32, nullptr, IN(bb + 4), R1, 0, IN(bb + 5),
                    nullptr, 0, F + o_dl1, D1, (int)L, D1, R1, 0);
        hipLaunchKernelGGL(scan_kernel, grid1d(D1, 64), dim3(64), 0, stream,
                           F + o_xc1, F + o_dl1, IN(bb + 6),
                           F + o_xd1 + R1, R1 + 32, F + o_xd1 + R1 + 16, R1 + 32,
                           IN(bb + 7), F + o_in1 + D1, 2 * D1, IN(bb + 5),
                           F + o_y1, D1, L);
        launch_gemm(stream, F + o_y1, D1, nullptr, IN(bb + 8), D1, 0, nullptr,
                    m_l, 256, m_l, 256, (int)L, DIM, D1, 0);

        // -------- STS block (dim=192, d=384, r=12) --------
        launch_gemm(stream, m_h, 256, nullptr, IN(sb + 0), DIM3, 0, nullptr,
                    nullptr, 0, F + o_in2, D2, (int)L, D2, DIM3, 0);
        hipLaunchKernelGGL(dwconv_silu_kernel, grid1d((long)D2 * L, 256), dim3(256), 0, stream,
                           F + o_in2, D2, IN(sb + 1), IN(sb + 2), F + o_xc2, D2, L);
        hipLaunchKernelGGL(ln_kernel, grid1d(L, 256), dim3(256), 0, stream,
                           kex, 192, IN(sb + 3), IN(sb + 4), F + o_lnb, DIM3, L);
        launch_gemm(stream, F + o_lnb, DIM3, nullptr, IN(sb + 5), DIM3, 0, IN(sb + 6),
                    nullptr, 0, F + o_kp, D2, (int)L, D2, DIM3, 1);
        hipLaunchKernelGGL(ln_kernel, grid1d(L, 256), dim3(256), 0, stream,
                           p_h, 256, IN(sb + 7), IN(sb + 8), F + o_lnb, DIM3, L);
        launch_gemm(stream, F + o_lnb, DIM3, nullptr, IN(sb + 9), DIM3, 0, IN(sb + 10),
                    nullptr, 0, F + o_qp, D2, (int)L, D2, DIM3, 1);
        // x_dbl = concat([x_t, Kp]) @ dt_bc_w  (two accumulating passes)
        launch_gemm(stream, F + o_xc2, D2, nullptr, IN(sb + 11), 2 * D2, 0, nullptr,
                    nullptr, 0, F + o_xd2, R2 + 32, (int)L, R2 + 32, D2, 0);
        launch_gemm(stream, F + o_kp, D2, nullptr, IN(sb + 11), 2 * D2, D2, nullptr,
                    F + o_xd2, R2 + 32, F + o_xd2, R2 + 32, (int)L, R2 + 32, D2, 0);
        // z_g = concat([x_t, Qp]) @ gate_w + gate_b  (two accumulating passes)
        launch_gemm(stream, F + o_xc2, D2, nullptr, IN(sb + 14), 2 * D2, 0, nullptr,
                    nullptr, 0, F + o_gt, D2, (int)L, D2, D2, 0);
        launch_gemm(stream, F + o_qp, D2, nullptr, IN(sb + 14), 2 * D2, D2, IN(sb + 15),
                    F + o_gt, D2, F + o_gt, D2, (int)L, D2, D2, 0);
        launch_gemm(stream, F + o_xd2, R2 + 32, nullptr, IN(sb + 12), R2, 0, IN(sb + 13),
                    nullptr, 0, F + o_dl2, D2, (int)L, D2, R2, 0);
        hipLaunchKernelGGL(scan_kernel, grid1d(D2, 64), dim3(64), 0, stream,
                           F + o_xc2, F + o_dl2, IN(sb + 16),
                           F + o_xd2 + R2, R2 + 32, F + o_xd2 + R2 + 16, R2 + 32,
                           IN(sb + 17), F + o_gt, D2, IN(sb + 13),
                           F + o_y2, D2, L);
        launch_gemm(stream, F + o_y2, D2, nullptr, IN(sb + 18), D2, 0, nullptr,
                    m_h, 256, m_h, 256, (int)L, DIM3, D2, 0);
    }

    // ================= IDWT + ADR + tail =================
    hipLaunchKernelGGL(idwt_kernel, grid1d(64L * L, 256), dim3(256), 0, stream,
                       m_d, F + o_idwt, 64, H2, W2);
    hipLaunchKernelGGL(chstats_kernel, dim3(64), dim3(256), 0, stream,
                       F + o_idwt, adry, 64, HW);
    hipLaunchKernelGGL(adr_gate_kernel, dim3(1), dim3(64), 0, stream,
                       adry, IN(100), adrg, 64);
    hipLaunchKernelGGL(scale_kernel, grid1d(64L * HW, 256), dim3(256), 0, stream,
                       F + o_idwt, adrg, 64, HW);
    hipLaunchKernelGGL(conv3x3_kernel, grid1d(8L * HW, 256), dim3(256), 0, stream,
                       F + o_idwt, IN(101), IN(102), ms, (float*)d_out, 64, 8, H, W, 0);
#undef IN
}
